// DriftRectifier_1116691496969
// MI455X (gfx1250) — matrix operand order repaired and checked by reading, not compiled, not run on hardware
//
#include <hip/hip_runtime.h>
#include <hip/hip_bf16.h>

// ---------------- problem constants ----------------
#define Bb   16
#define Cc   4
#define Hh   64
#define Www  64
#define Ll   (Hh * Www)       // 4096
#define BL   (Bb * Ll)        // 65536
#define DM   64
#define DI   128
#define DS_  16
#define DC   4
#define DR   4
#define CHUNKS 32
#define CLEN (Ll / CHUNKS)    // 128

typedef __attribute__((ext_vector_type(16))) _Float16 v16h;
typedef __attribute__((ext_vector_type(8)))  float    v8f;

__device__ __forceinline__ float sigf(float x) { return 1.0f / (1.0f + __expf(-x)); }
__device__ __forceinline__ float siluf(float x) { return x * sigf(x); }

// ======================================================================
// WMMA GEMM:  C[M,N] = A[M,K] @ W[N,K]^T
// K compile-time (64/128) -> fully unrolled WMMA chain.
// Each wave owns one 16-row tile and NT consecutive 16-col tiles,
// reusing the converted A fragments across all NT column tiles.
// OOB columns: clamped row index + multiplicative zero mask (branchless,
// EXEC stays all-1s through every WMMA).
// ======================================================================
template <int K, int NT>
__global__ void __launch_bounds__(128)
gemm_wmma_nt(const float* __restrict__ A, int lda,
             const float* __restrict__ W, int ldw,
             float* __restrict__ C, int ldc, int N)
{
    constexpr int KS = K / 32;
    const int lane = threadIdx.x & 31;
    const int wave = threadIdx.x >> 5;
    const int n0   = (blockIdx.y * 4 + wave) * (16 * NT);
    if (n0 >= N) return;                        // wave-uniform
    const int row0 = blockIdx.x * 16;

    const int hh = lane >> 4;                   // half-wave select
    const int r  = lane & 15;

    const float* Arow = A + (size_t)(row0 + r) * lda;

    // Preload + convert all A fragments (ISA 16-bit A layout:
    // lanes<16: K {2v,2v+1} / {16+2(v-4)+...}; hi half +8)
    v16h afr[KS];
#pragma unroll
    for (int ks = 0; ks < KS; ++ks) {
        const int kk = ks * 32;
#pragma unroll
        for (int v = 0; v < 8; ++v) {
            int kb = (v < 4) ? (2 * v + (hh ? 8 : 0))
                             : (16 + 2 * (v - 4) + (hh ? 8 : 0));
            afr[ks][2 * v]     = (_Float16)Arow[kk + kb];
            afr[ks][2 * v + 1] = (_Float16)Arow[kk + kb + 1];
        }
    }

#pragma unroll
    for (int t = 0; t < NT; ++t) {
        const int ncol  = n0 + t * 16 + r;
        const int nclmp = (ncol < N) ? ncol : (N - 1);
        const float msk = (ncol < N) ? 1.0f : 0.0f;
        const float* Wrow = W + (size_t)nclmp * ldw;

        v8f acc = {};
#pragma unroll
        for (int ks = 0; ks < KS; ++ks) {
            const float* Wk = Wrow + ks * 32 + (hh ? 8 : 0);
            v16h bf;
#pragma unroll
            for (int v = 0; v < 8; ++v) {
                const int kb = (v < 4) ? (2 * v) : (2 * v + 8);
                bf[2 * v]     = (_Float16)(msk * Wk[kb]);
                bf[2 * v + 1] = (_Float16)(msk * Wk[kb + 1]);
            }
            acc = __builtin_amdgcn_wmma_f32_16x16x32_f16(false, afr[ks], false, bf,
                                                         (short)0, acc, false, false);
        }

        if (ncol < N) {
#pragma unroll
            for (int v = 0; v < 8; ++v) {
                int row = row0 + v + 8 * hh;    // C/D layout: VGPR v -> M = v + 8*half
                C[(size_t)row * ldc + ncol] = acc[v];
            }
        }
    }
}

// ======================================================================
// embed: feat[b,l,dm] = sum_c z[b,c,l]*emb_w[dm,c] + emb_b[dm]
// ======================================================================
__global__ void __launch_bounds__(256)
embed_kernel(const float* __restrict__ z, const float* __restrict__ ew,
             const float* __restrict__ eb, float* __restrict__ feat)
{
    size_t idx = (size_t)blockIdx.x * 256 + threadIdx.x;   // over BL*64
    int dm = idx & 63;
    size_t bl = idx >> 6;
    int b = (int)(bl >> 12);
    int l = (int)(bl & 4095);
    float acc = eb[dm];
#pragma unroll
    for (int c = 0; c < Cc; ++c)
        acc += z[(((size_t)(b * Cc + c)) << 12) + l] * ew[dm * Cc + c];
    feat[idx] = acc;
}

// ======================================================================
// causal depthwise conv (K=4) over L + bias + SiLU.  xz row = 256 floats.
// ======================================================================
__global__ void __launch_bounds__(256)
conv_silu_kernel(const float* __restrict__ xz, const float* __restrict__ cw,
                 const float* __restrict__ cb, float* __restrict__ u)
{
    size_t idx = (size_t)blockIdx.x * 256 + threadIdx.x;   // over BL*128
    int d = idx & 127;
    size_t bl = idx >> 7;
    int l = (int)(bl & 4095);
    size_t rowbase = bl - (size_t)l;                       // b*L
    float acc = cb[d];
#pragma unroll
    for (int k = 0; k < DC; ++k) {
        int lk = l - (DC - 1) + k;
        if (lk >= 0)
            acc += cw[d * DC + k] * xz[(rowbase + lk) * 256 + d];
    }
    u[idx] = siluf(acc);
}

// ======================================================================
// dt_full = softplus(dt_raw @ dt_proj_w^T + dt_proj_b); xdbl row = 40 floats
// ======================================================================
__global__ void __launch_bounds__(256)
dtproj_kernel(const float* __restrict__ xd, const float* __restrict__ dtw,
              const float* __restrict__ dtb, float* __restrict__ dt)
{
    size_t idx = (size_t)blockIdx.x * 256 + threadIdx.x;   // over BL*128
    int d = idx & 127;
    size_t bl = idx >> 7;
    const float* xr = xd + bl * 40;
    float acc = dtb[d];
#pragma unroll
    for (int r = 0; r < DR; ++r) acc += xr[r] * dtw[d * DR + r];
    dt[idx] = (acc > 20.0f) ? acc : log1pf(__expf(acc));
}

// ======================================================================
// chunked scan, phase 1: per chunk reduce affine maps -> (P = prod dA, S)
// thread = (d within 16-chunk, n); grid = (DI/16, B, CHUNKS)
// ======================================================================
__global__ void __launch_bounds__(256)
scan_pass1(const float* __restrict__ dt, const float* __restrict__ u,
           const float* __restrict__ xd, const float* __restrict__ Alog,
           float* __restrict__ P, float* __restrict__ S)
{
    int t = threadIdx.x;
    int d = blockIdx.x * 16 + (t >> 4);
    int n = t & 15;
    int b = blockIdx.y, ch = blockIdx.z;
    float a_dn = -__expf(Alog[d * DS_ + n]);
    float p = 1.0f, s = 0.0f;
    size_t base = (size_t)b * Ll + (size_t)ch * CLEN;
    for (int i = 0; i < CLEN; ++i) {
        size_t bl = base + i;
        float dtv = dt[bl * DI + d];
        float uv  = u [bl * DI + d];
        float Bv  = xd[bl * 40 + DR + n];
        float dA  = __expf(dtv * a_dn);
        s = dA * s + dtv * Bv * uv;
        p = dA * p;
    }
    size_t o = (((size_t)b * CHUNKS + ch) * DI + d) * DS_ + n;
    P[o] = p; S[o] = s;
}

// phase 2: sequential combine over 32 chunks -> chunk start states
__global__ void __launch_bounds__(256)
scan_pass2(const float* __restrict__ P, const float* __restrict__ S,
           float* __restrict__ Hst)
{
    int t = threadIdx.x;
    int d = blockIdx.x * 16 + (t >> 4);
    int n = t & 15;
    int b = blockIdx.y;
    float h = 0.0f;
    for (int ch = 0; ch < CHUNKS; ++ch) {
        size_t o = (((size_t)b * CHUNKS + ch) * DI + d) * DS_ + n;
        Hst[o] = h;
        h = P[o] * h + S[o];
    }
}

// phase 3: re-scan each chunk from its start state, emit y (+ u*D)
__global__ void __launch_bounds__(256)
scan_pass3(const float* __restrict__ dt, const float* __restrict__ u,
           const float* __restrict__ xd, const float* __restrict__ Alog,
           const float* __restrict__ Dp, const float* __restrict__ Hst,
           float* __restrict__ y)
{
    int t = threadIdx.x;
    int d = blockIdx.x * 16 + (t >> 4);
    int n = t & 15;
    int b = blockIdx.y, ch = blockIdx.z;
    float a_dn = -__expf(Alog[d * DS_ + n]);
    float dval = Dp[d];
    float h = Hst[(((size_t)b * CHUNKS + ch) * DI + d) * DS_ + n];
    size_t base = (size_t)b * Ll + (size_t)ch * CLEN;
    for (int i = 0; i < CLEN; ++i) {
        size_t bl = base + i;
        float dtv = dt[bl * DI + d];
        float uv  = u [bl * DI + d];
        float Bv  = xd[bl * 40 + DR + n];
        float Cv  = xd[bl * 40 + DR + DS_ + n];
        h = __expf(dtv * a_dn) * h + dtv * Bv * uv;
        float p = h * Cv;
#pragma unroll
        for (int m = 1; m < 16; m <<= 1) p += __shfl_xor(p, m, 32);
        if (n == 0) y[bl * DI + d] = p + uv * dval;
    }
}

// gating: y *= silu(z), z = xz[...,128:256]
__global__ void __launch_bounds__(256)
gate_kernel(const float* __restrict__ xz, float* __restrict__ y)
{
    size_t idx = (size_t)blockIdx.x * 256 + threadIdx.x;   // over BL*128
    int d = idx & 127;
    size_t bl = idx >> 7;
    y[idx] *= siluf(xz[bl * 256 + 128 + d]);
}

// LayerNorm over rows of 64; one wave per row (2 elems/lane), in place
__global__ void __launch_bounds__(256)
layernorm_kernel(float* __restrict__ X, const float* __restrict__ g,
                 const float* __restrict__ bta)
{
    int lane = threadIdx.x & 31;
    int wave = threadIdx.x >> 5;
    size_t row = (size_t)blockIdx.x * 8 + wave;
    float* xr = X + row * DM;
    float x0 = xr[lane], x1 = xr[lane + 32];
    float s = x0 + x1;
#pragma unroll
    for (int m = 16; m >= 1; m >>= 1) s += __shfl_xor(s, m, 32);
    float mu = s * (1.0f / 64.0f);
    float d0 = x0 - mu, d1 = x1 - mu;
    float v = d0 * d0 + d1 * d1;
#pragma unroll
    for (int m = 16; m >= 1; m >>= 1) v += __shfl_xor(v, m, 32);
    float rs = rsqrtf(v * (1.0f / 64.0f) + 1e-5f);
    xr[lane]      = d0 * rs * g[lane]      + bta[lane];
    xr[lane + 32] = d1 * rs * g[lane + 32] + bta[lane + 32];
}

// head + residual + NHWC->NCHW: out[b,c,l] = z[b,c,l] - (feat[b,l,:]@hw[c,:]+hb[c])
__global__ void __launch_bounds__(256)
head_kernel(const float* __restrict__ feat, const float* __restrict__ hw,
            const float* __restrict__ hb, const float* __restrict__ z,
            float* __restrict__ out)
{
    size_t idx = (size_t)blockIdx.x * 256 + threadIdx.x;   // over BL*4
    int c = idx & 3;
    size_t bl = idx >> 2;
    int b = (int)(bl >> 12);
    int l = (int)(bl & 4095);
    const float* f = feat + bl * DM;
    const float* w = hw + c * DM;
    float acc = hb[c];
#pragma unroll 8
    for (int d = 0; d < DM; ++d) acc += f[d] * w[d];
    size_t o = (((size_t)(b * Cc + c)) << 12) + l;
    out[o] = z[o] - acc;
}

// ======================================================================
// host-side orchestration
// ======================================================================
struct MambaP {
    const float *inp, *cw, *cb, *xp, *dtw, *dtb, *alog, *dp, *outp;
};

struct WsBuf {
    float *xz, *u, *xd, *dt, *y, *P, *S, *Hst;
};

static void run_mamba(const float* x, float* out, const MambaP& w, const WsBuf& ws,
                      hipStream_t stream)
{
    // in_proj: [BL,64] x [256,64]^T -> xz [BL,256]; 4 waves x 4 tiles cover N=256
    gemm_wmma_nt<64, 4><<<dim3(BL / 16, 1), 128, 0, stream>>>(x, DM, w.inp, DM,
                                                              ws.xz, 2 * DI, 2 * DI);
    // causal conv + SiLU -> u [BL,128]
    conv_silu_kernel<<<(BL * DI) / 256, 256, 0, stream>>>(ws.xz, w.cw, w.cb, ws.u);
    // x_proj: [BL,128] x [36,128]^T -> xd [BL,40-padded]
    gemm_wmma_nt<128, 1><<<dim3(BL / 16, 1), 128, 0, stream>>>(ws.u, DI, w.xp, DI,
                                                               ws.xd, 40, DR + 2 * DS_);
    // dt_proj + softplus -> dt [BL,128]
    dtproj_kernel<<<(BL * DI) / 256, 256, 0, stream>>>(ws.xd, w.dtw, w.dtb, ws.dt);
    // chunked selective scan
    scan_pass1<<<dim3(DI / 16, Bb, CHUNKS), 256, 0, stream>>>(ws.dt, ws.u, ws.xd,
                                                              w.alog, ws.P, ws.S);
    scan_pass2<<<dim3(DI / 16, Bb), 256, 0, stream>>>(ws.P, ws.S, ws.Hst);
    scan_pass3<<<dim3(DI / 16, Bb, CHUNKS), 256, 0, stream>>>(ws.dt, ws.u, ws.xd,
                                                              w.alog, w.dp, ws.Hst, ws.y);
    // gate: y *= silu(z)
    gate_kernel<<<(BL * DI) / 256, 256, 0, stream>>>(ws.xz, ws.y);
    // out_proj: [BL,128] x [64,128]^T -> out [BL,64]
    gemm_wmma_nt<128, 1><<<dim3(BL / 16, 1), 128, 0, stream>>>(ws.y, DI, w.outp, DI,
                                                               out, DM, DM);
}

extern "C" void kernel_launch(void* const* d_in, const int* in_sizes, int n_in,
                              void* d_out, int out_size, void* d_ws, size_t ws_size,
                              hipStream_t stream)
{
    const float* z     = (const float*)d_in[0];
    const float* emb_w = (const float*)d_in[1];
    const float* emb_b = (const float*)d_in[2];
    MambaP m1 { (const float*)d_in[3],  (const float*)d_in[4],  (const float*)d_in[5],
                (const float*)d_in[6],  (const float*)d_in[7],  (const float*)d_in[8],
                (const float*)d_in[9],  (const float*)d_in[10], (const float*)d_in[11] };
    MambaP m2 { (const float*)d_in[12], (const float*)d_in[13], (const float*)d_in[14],
                (const float*)d_in[15], (const float*)d_in[16], (const float*)d_in[17],
                (const float*)d_in[18], (const float*)d_in[19], (const float*)d_in[20] };
    const float* ln1_g = (const float*)d_in[21];
    const float* ln1_b = (const float*)d_in[22];
    const float* ln2_g = (const float*)d_in[23];
    const float* ln2_b = (const float*)d_in[24];
    const float* hw    = (const float*)d_in[25];
    const float* hb    = (const float*)d_in[26];
    float* out = (float*)d_out;

    float* f = (float*)d_ws;
    float* bufA  = f;                   f += (size_t)BL * DM;      // feat
    float* bufF2 = f;                   f += (size_t)BL * DM;      // feat2
    WsBuf ws;
    ws.xz  = f;                         f += (size_t)BL * 2 * DI;
    ws.u   = f;                         f += (size_t)BL * DI;
    ws.xd  = f;                         f += (size_t)BL * 40;
    ws.dt  = f;                         f += (size_t)BL * DI;
    ws.y   = f;                         f += (size_t)BL * DI;
    ws.P   = f;                         f += (size_t)Bb * CHUNKS * DI * DS_;
    ws.S   = f;                         f += (size_t)Bb * CHUNKS * DI * DS_;
    ws.Hst = f;                         f += (size_t)Bb * CHUNKS * DI * DS_;

    // embedding: z [B,C,H,W] -> feat [B*L, 64]
    embed_kernel<<<(BL * DM) / 256, 256, 0, stream>>>(z, emb_w, emb_b, bufA);

    run_mamba(bufA, bufF2, m1, ws, stream);
    layernorm_kernel<<<BL / 8, 256, 0, stream>>>(bufF2, ln1_g, ln1_b);

    run_mamba(bufF2, bufA, m2, ws, stream);
    layernorm_kernel<<<BL / 8, 256, 0, stream>>>(bufA, ln2_g, ln2_b);

    // head + residual + transpose back to NCHW
    head_kernel<<<(BL * Cc) / 256, 256, 0, stream>>>(bufA, hw, hb, z, out);
}
